// Sequence_45964740001905
// MI455X (gfx1250) — matrix operand order repaired and checked by reading, not compiled, not run on hardware
//
#include <hip/hip_runtime.h>
#include <hip/hip_bf16.h>

// ---------------- problem constants (from reference) ----------------
#define HID       101
#define G4        404            // 4*HID gate width
#define T_IN      256
#define B_TOTAL   1024
#define NT        26             // ceil(404/16) N-tiles
#define NTHREADS  (NT * 32)      // 832 threads = 26 wave32 waves
#define K1PAD     128            // K for layer1: 3 + 101 = 104 -> pad 128
#define K2PAD     224            // K for layer2: 101 + 101 = 202 -> pad 224
#define GSTRT     20             // transposed gate buffer row stride (floats)

typedef __attribute__((ext_vector_type(16))) __bf16 v16bf;
typedef __attribute__((ext_vector_type(8)))  __bf16 v8bf;
typedef __attribute__((ext_vector_type(8)))  float  v8f;
typedef __attribute__((ext_vector_type(4)))  float  v4f;

union AFrag { v16bf v; v8bf h[2]; };

// fast transcendentals: v_exp_f32 + v_rcp_f32, no IEEE-division expansion
__device__ __forceinline__ float fast_sigmoid(float x) {
    return __builtin_amdgcn_rcpf(1.f + __expf(-x));
}
__device__ __forceinline__ float fast_tanh(float x) {
    return 2.f * __builtin_amdgcn_rcpf(1.f + __expf(-2.f * x)) - 1.f;
}

// Each workgroup owns 16 batch rows: a fully independent 2-layer LSTM
// recurrence. 26 waves each own one 16-wide column tile of the 404-wide gate
// GEMM; weight B-operand fragments live in VGPRs for the whole kernel.
__global__ void __launch_bounds__(NTHREADS, 1)
lstm2_wmma_kernel(const float* __restrict__ input,
                  const float* __restrict__ Wih1, const float* __restrict__ Whh1,
                  const float* __restrict__ bih1, const float* __restrict__ bhh1,
                  const float* __restrict__ Wih2, const float* __restrict__ Whh2,
                  const float* __restrict__ bih2, const float* __restrict__ bhh2,
                  const float* __restrict__ Wl,   const float* __restrict__ bl,
                  const int*   __restrict__ futurep,
                  float* __restrict__ out)
{
    __shared__ __align__(16) __bf16 sA1[16][K1PAD];     // [x_t | h1]
    __shared__ __align__(16) __bf16 sA2[16][K2PAD];     // [h1  | h2]
    __shared__ __align__(16) float  sGateT[416][GSTRT]; // activated gates, [col][row]
    __shared__ float sC1[16][HID];
    __shared__ float sC2[16][HID];
    __shared__ float sWl[3][HID];
    __shared__ float sBl[3];

    const int tid   = threadIdx.x;
    const int lane  = tid & 31;
    const int wave  = tid >> 5;        // N-tile id 0..25
    const int n     = lane & 15;
    const int hi    = lane >> 4;       // half-wave select
    const int gcol  = wave * 16 + n;   // gate column this lane produces (0..415)
    const int bkoff = hi << 4;         // unused; B-operand K offset is akoff (0/8)
    const int am    = n;               // A-operand matrix row (batch row in tile)
    const int akoff = hi << 3;         // A-operand K offset inside 32-chunk (0/8)
    const int mbase = hi << 3;         // C/D row base (0/8)
    const int row0  = blockIdx.x * 16; // first batch row of this workgroup

    // ---- zero recurrent state + padding, stage projection weights ----
    for (int k = tid; k < 16 * K1PAD; k += NTHREADS) ((__bf16*)sA1)[k] = (__bf16)0.f;
    for (int k = tid; k < 16 * K2PAD; k += NTHREADS) ((__bf16*)sA2)[k] = (__bf16)0.f;
    for (int k = tid; k < 16 * HID;   k += NTHREADS) { ((float*)sC1)[k] = 0.f; ((float*)sC2)[k] = 0.f; }
    for (int k = tid; k < 3 * HID;    k += NTHREADS) sWl[k / HID][k % HID] = Wl[k];
    if (tid < 3) sBl[tid] = bl[tid];

    // ---- resident bf16 weight fragments (B operand) -------------------
    // Layer1 fused K axis: [W_ih1 (K=0..2) | W_hh1 (K=3..103) | 0-pad]
    v16bf w1[4];
#pragma unroll
    for (int kc = 0; kc < 4; ++kc) {
        v16bf f;
#pragma unroll
        for (int e = 0; e < 16; ++e) {
            int K = kc * 32 + akoff + e + (e >= 8 ? 8 : 0);
            float v = 0.f;
            if (gcol < G4) {
                if (K < 3)            v = Wih1[gcol * 3 + K];
                else if (K < 3 + HID) v = Whh1[gcol * HID + (K - 3)];
            }
            f[e] = (__bf16)v;
        }
        w1[kc] = f;
    }
    // Layer2 fused K axis: [W_ih2 (K=0..100) | W_hh2 (K=101..201) | 0-pad]
    v16bf w2[7];
#pragma unroll
    for (int kc = 0; kc < 7; ++kc) {
        v16bf f;
#pragma unroll
        for (int e = 0; e < 16; ++e) {
            int K = kc * 32 + akoff + e + (e >= 8 ? 8 : 0);
            float v = 0.f;
            if (gcol < G4) {
                if (K < HID)          v = Wih2[gcol * HID + K];
                else if (K < 2 * HID) v = Whh2[gcol * HID + (K - HID)];
            }
            f[e] = (__bf16)v;
        }
        w2[kc] = f;
    }

    float bias1 = 0.f, bias2 = 0.f;
    if (gcol < G4) {
        bias1 = bih1[gcol] + bhh1[gcol];
        bias2 = bih2[gcol] + bhh2[gcol];
    }
    const bool isG = (gcol >= 2 * HID) && (gcol < 3 * HID);   // tanh gate column?

    const int Ttot = T_IN + futurep[0];

    // prime the software-pipelined x register (x for t = 0)
    const int xm = tid / 3, xe = tid % 3;     // valid for tid < 48
    float xr = 0.f;
    if (tid < 48) xr = input[((long long)(row0 + xm) * T_IN) * 3 + xe];
    __syncthreads();

    for (int t = 0; t < Ttot; ++t) {
        // stage current x (teacher-forced phase); autoregressive phase reuses
        // the projection feedback already written into sA1[:,0..2]
        if (t < T_IN && tid < 48) sA1[xm][xe] = (__bf16)xr;
        __syncthreads();

        // issue next step's x load now: completes during GEMM2/elementwise
        float xnext = 0.f;
        if (t + 1 < T_IN && tid < 48)
            xnext = input[((long long)(row0 + xm) * T_IN + (t + 1)) * 3 + xe];

        // ---------------- layer-1 gate GEMM (K = 128) ----------------
        {
            AFrag a[4];
#pragma unroll
            for (int kc = 0; kc < 4; ++kc) {     // all loads issue up front
                a[kc].h[0] = *(const v8bf*)&sA1[am][kc * 32 + akoff];
                a[kc].h[1] = *(const v8bf*)&sA1[am][kc * 32 + 16 + akoff];
            }
            v8f acc = {bias1, bias1, bias1, bias1, bias1, bias1, bias1, bias1};
#pragma unroll
            for (int kc = 0; kc < 4; ++kc)
                acc = __builtin_amdgcn_wmma_f32_16x16x32_bf16(
                          false, a[kc].v, false, w1[kc], (short)0, acc, false, false);
            if (gcol < G4) {
                v4f lo, hi;
#pragma unroll
                for (int r = 0; r < 4; ++r) {
                    float s0 = fast_sigmoid(isG ? 2.f * acc[r]     : acc[r]);
                    float s1 = fast_sigmoid(isG ? 2.f * acc[r + 4] : acc[r + 4]);
                    lo[r] = isG ? (2.f * s0 - 1.f) : s0;   // tanh via sigmoid
                    hi[r] = isG ? (2.f * s1 - 1.f) : s1;
                }
                *(v4f*)&sGateT[gcol][mbase]     = lo;   // rows mbase..mbase+3
                *(v4f*)&sGateT[gcol][mbase + 4] = hi;   // rows mbase+4..mbase+7
            }
        }
        __syncthreads();

        // ---------------- layer-1 cell update (fp32 c) ----------------
        for (int idx = tid; idx < 16 * HID; idx += NTHREADS) {
            int m = idx / HID, j = idx % HID;
            float ig = sGateT[j][m],           fg = sGateT[HID + j][m];
            float gg = sGateT[2 * HID + j][m], og = sGateT[3 * HID + j][m];
            float c  = fg * sC1[m][j] + ig * gg;
            sC1[m][j] = c;
            __bf16 hb = (__bf16)(og * fast_tanh(c));
            sA2[m][j]     = hb;   // layer-2 input (this step)
            sA1[m][3 + j] = hb;   // layer-1 recurrent input (next step)
        }
        __syncthreads();

        // ---------------- layer-2 gate GEMM (K = 224) ----------------
        {
            AFrag a[7];
#pragma unroll
            for (int kc = 0; kc < 7; ++kc) {
                a[kc].h[0] = *(const v8bf*)&sA2[am][kc * 32 + akoff];
                a[kc].h[1] = *(const v8bf*)&sA2[am][kc * 32 + 16 + akoff];
            }
            v8f acc = {bias2, bias2, bias2, bias2, bias2, bias2, bias2, bias2};
#pragma unroll
            for (int kc = 0; kc < 7; ++kc)
                acc = __builtin_amdgcn_wmma_f32_16x16x32_bf16(
                          false, a[kc].v, false, w2[kc], (short)0, acc, false, false);
            if (gcol < G4) {
                v4f lo, hi;
#pragma unroll
                for (int r = 0; r < 4; ++r) {
                    float s0 = fast_sigmoid(isG ? 2.f * acc[r]     : acc[r]);
                    float s1 = fast_sigmoid(isG ? 2.f * acc[r + 4] : acc[r + 4]);
                    lo[r] = isG ? (2.f * s0 - 1.f) : s0;
                    hi[r] = isG ? (2.f * s1 - 1.f) : s1;
                }
                *(v4f*)&sGateT[gcol][mbase]     = lo;
                *(v4f*)&sGateT[gcol][mbase + 4] = hi;
            }
        }
        __syncthreads();

        // ---------------- layer-2 cell update ----------------
        for (int idx = tid; idx < 16 * HID; idx += NTHREADS) {
            int m = idx / HID, j = idx % HID;
            float ig = sGateT[j][m],           fg = sGateT[HID + j][m];
            float gg = sGateT[2 * HID + j][m], og = sGateT[3 * HID + j][m];
            float c  = fg * sC2[m][j] + ig * gg;
            sC2[m][j] = c;
            sA2[m][HID + j] = (__bf16)(og * fast_tanh(c));  // h2: next step + projection
        }
        __syncthreads();

        // -------- output projection h2 @ Wl^T + bl (16 lanes per output) --------
        if (tid < 768) {
            int oid = tid >> 4;              // 0..47
            int l   = tid & 15;
            int m   = oid / 3, e = oid % 3;
            float sum = 0.f;
#pragma unroll
            for (int j = l; j < HID; j += 16)
                sum += (float)sA2[m][HID + j] * sWl[e][j];
#pragma unroll
            for (int off = 8; off > 0; off >>= 1)
                sum += __shfl_xor(sum, off, 16);
            if (l == 0) {
                sum += sBl[e];
                out[((long long)(row0 + m) * Ttot + t) * 3 + e] = sum;
                if (t >= T_IN - 1)          // feed back as next x in future phase
                    sA1[m][e] = (__bf16)sum;
            }
        }
        xr = xnext;
        // loop-top __syncthreads() orders projection/feedback vs next GEMM1
    }
}

extern "C" void kernel_launch(void* const* d_in, const int* in_sizes, int n_in,
                              void* d_out, int out_size, void* d_ws, size_t ws_size,
                              hipStream_t stream) {
    const float* input = (const float*)d_in[0];
    const float* Wih1  = (const float*)d_in[1];
    const float* Whh1  = (const float*)d_in[2];
    const float* bih1  = (const float*)d_in[3];
    const float* bhh1  = (const float*)d_in[4];
    const float* Wih2  = (const float*)d_in[5];
    const float* Whh2  = (const float*)d_in[6];
    const float* bih2  = (const float*)d_in[7];
    const float* bhh2  = (const float*)d_in[8];
    const float* Wl    = (const float*)d_in[9];
    const float* bl    = (const float*)d_in[10];
    const int*   fut   = (const int*)d_in[11];
    float* outp = (float*)d_out;

    dim3 grid(B_TOTAL / 16);   // 64 independent 16-row recurrences
    dim3 block(NTHREADS);      // 26 waves <-> 26 gate-column tiles
    hipLaunchKernelGGL(lstm2_wmma_kernel, grid, block, 0, stream,
                       input, Wih1, Whh1, bih1, bhh1,
                       Wih2, Whh2, bih2, bhh2, Wl, bl, fut, outp);
}
